// Controller_53575422050930
// MI455X (gfx1250) — matrix operand order repaired and checked by reading, not compiled, not run on hardware
//
#include <hip/hip_runtime.h>
#include <hip/hip_bf16.h>
#include <math.h>

#define H_      2048
#define G4_     8192
#define NOPS_   8
#define NCMP_   6
#define NANCH_  8
#define NEGF    (-1e9f)
#define NBLOCKS 64
#define NTHREADS 256

typedef __attribute__((ext_vector_type(16))) __bf16 v16bf;
typedef __attribute__((ext_vector_type(8)))  float  v8f;

// ---------------- workspace layout (bytes) ----------------
static constexpr size_t align_up(size_t x, size_t a) { return (x + a - 1) & ~(a - 1); }
static constexpr size_t OFF_BAR   = 0;                                     // 2 x uint barrier
static constexpr size_t OFF_SCAL  = 64;                                    // lp, ent
static constexpr size_t OFF_STAGE = 128;                                   // 48 floats (idx/opt staging)
static constexpr size_t OFF_G     = align_up(OFF_STAGE + 48 * 4, 256);     // gates [8192]
static constexpr size_t OFF_H     = OFF_G    + (size_t)G4_ * 4;            // h [2048]
static constexpr size_t OFF_C     = OFF_H    + (size_t)H_ * 4;             // c [2048]
static constexpr size_t OFF_T     = OFF_C    + (size_t)H_ * 4;             // t = h@w2.T [2048]
static constexpr size_t OFF_ANCH  = OFF_T    + (size_t)H_ * 4;             // anchors [8][2048]
static constexpr size_t OFF_AW1   = OFF_ANCH + (size_t)NANCH_ * H_ * 4;    // anchors_w1 [8][2048]
static constexpr size_t OFF_XBF   = align_up(OFF_AW1 + (size_t)NANCH_ * H_ * 4, 256); // x bf16
static constexpr size_t OFF_HBF   = OFF_XBF  + (size_t)H_ * 2;             // h bf16
static constexpr size_t OFF_EMBBF = OFF_HBF  + (size_t)H_ * 2;             // emb bf16 [9][2048]
static constexpr size_t OFF_WIH   = align_up(OFF_EMBBF + (size_t)(NOPS_ + 1) * H_ * 2, 256);
static constexpr size_t OFF_WHH   = OFF_WIH  + (size_t)G4_ * H_ * 2;
static constexpr size_t OFF_W1    = OFF_WHH  + (size_t)G4_ * H_ * 2;
static constexpr size_t OFF_W2    = OFF_W1   + (size_t)H_ * H_ * 2;

// ---------------- helpers ----------------
__device__ __forceinline__ unsigned short f2bf(float f) {
  unsigned u = __float_as_uint(f);
  unsigned r = ((u >> 16) & 1u) + 0x7FFFu;  // round-to-nearest-even
  return (unsigned short)((u + r) >> 16);
}

__device__ __forceinline__ v16bf ld_b16(const unsigned short* p) {
  union { v16bf v; uint4 q[2]; } u;
  u.q[0] = *reinterpret_cast<const uint4*>(p);
  u.q[1] = *reinterpret_cast<const uint4*>(p + 8);
  return u.v;
}
__device__ __forceinline__ v16bf ld_b16_split(const unsigned short* p0, const unsigned short* p1) {
  union { v16bf v; uint4 q[2]; } u;
  u.q[0] = *reinterpret_cast<const uint4*>(p0);
  u.q[1] = *reinterpret_cast<const uint4*>(p1);
  return u.v;
}

// WGP-scope prefetch (scope 0 pulls into all cache levels, incl. WGP$).
// __builtin_prefetch can only reach SE/SYS scope, which skip the WGP cache.
__device__ __forceinline__ void prefetch_wgp(const void* p) {
  unsigned long long a = (unsigned long long)p;
  asm volatile("global_prefetch_b8 %0, off offset:0" :: "v"(a));
}

__device__ __forceinline__ float sigmoidf_(float x) { return 1.0f / (1.0f + __expf(-x)); }

__device__ __forceinline__ unsigned mix32(unsigned a) {
  a ^= a >> 16; a *= 0x7feb352dU; a ^= a >> 15; a *= 0x846ca68bU; a ^= a >> 16; return a;
}
__device__ __forceinline__ float gumbel_from(unsigned h) {
  float u = ((float)(h >> 8) + 0.5f) * (1.0f / 16777216.0f);  // (0,1)
  return -logf(-logf(u));
}

// grid-wide sense barrier in workspace (all NBLOCKS blocks resident)
__device__ void grid_sync(unsigned* bar) {
  __syncthreads();
  if (threadIdx.x == 0) {
    __threadfence();
    unsigned gen = __atomic_load_n(bar + 1, __ATOMIC_ACQUIRE);
    if (atomicAdd(bar, 1u) == (unsigned)(NBLOCKS - 1)) {
      __atomic_store_n(bar, 0u, __ATOMIC_RELAXED);
      __threadfence();
      __atomic_fetch_add(bar + 1, 1u, __ATOMIC_RELEASE);
    } else {
      while (__atomic_load_n(bar + 1, __ATOMIC_ACQUIRE) == gen) {
        __builtin_amdgcn_s_sleep(2);
      }
    }
  }
  __syncthreads();
  __threadfence();
}

__device__ float block_reduce(float v, float* sred) {
  int t = threadIdx.x;
  sred[t] = v;
  __syncthreads();
  for (int off = NTHREADS / 2; off > 0; off >>= 1) {
    if (t < off) sred[t] += sred[t + off];
    __syncthreads();
  }
  float r = sred[0];
  __syncthreads();
  return r;
}

// ---------------- WMMA GEMV cores ----------------
// g[n] = sum_k x[k]*Wih[n,k] + sum_k h[k]*Whh[n,k], n in [0,8192).
// Wave w (512 total) owns rows nbase..nbase+15. A rows are all replicated with
// the vector chunk so every C row holds the result; B holds 16 weight rows.
// Two independent accumulators (one per matrix) so alternating WMMAs have no
// RAW dependency through the C/D registers.
__device__ void gemv_dual_wmma(float* g, const unsigned short* xbf, const unsigned short* hbf,
                               const unsigned short* wih, const unsigned short* whh) {
  int tid   = threadIdx.x;
  int gwave = (blockIdx.x * NTHREADS + tid) >> 5;
  int lane  = tid & 31;
  int nbase = gwave * 16;
  int col   = lane & 15;
  int hi    = lane >> 4;
  int ksel  = hi << 4;  // unused; B also uses asel
  int asel  = hi << 3;  // A: 0 or 8
  const unsigned short* rih = wih + (size_t)(nbase + col) * H_;
  const unsigned short* rhh = whh + (size_t)(nbase + col) * H_;
  v8f acc_ih = {};
  v8f acc_hh = {};
  for (int k0 = 0; k0 < H_; k0 += 32) {
    v16bf ax = ld_b16_split(xbf + k0 + asel, xbf + k0 + 16 + asel);
    v16bf bx = ld_b16_split(rih + k0 + asel, rih + k0 + 16 + asel);
    acc_ih = __builtin_amdgcn_wmma_f32_16x16x32_bf16(false, ax, false, bx, (short)0, acc_ih, false, false);
    v16bf ah = ld_b16_split(hbf + k0 + asel, hbf + k0 + 16 + asel);
    v16bf bh = ld_b16_split(rhh + k0 + asel, rhh + k0 + 16 + asel);
    acc_hh = __builtin_amdgcn_wmma_f32_16x16x32_bf16(false, ah, false, bh, (short)0, acc_hh, false, false);
    prefetch_wgp(rih + k0 + 1024);  // WGP-scope prefetch ~32 iters ahead
    prefetch_wgp(rhh + k0 + 1024);
  }
  if (lane < 16) g[nbase + lane] = acc_ih[0] + acc_hh[0];
}

// dst[n] = sum_k v[k]*W[n,k], n in [0, nout). Two K-interleaved accumulators.
__device__ void gemv_single_wmma(float* dst, const unsigned short* vbf,
                                 const unsigned short* w, int nout) {
  int tid   = threadIdx.x;
  int gwave = (blockIdx.x * NTHREADS + tid) >> 5;
  int lane  = tid & 31;
  int nbase = gwave * 16;
  if (nbase >= nout) return;
  int col  = lane & 15;
  int hi   = lane >> 4;
  int ksel = hi << 4;
  int asel = hi << 3;
  const unsigned short* rw = w + (size_t)(nbase + col) * H_;
  v8f acc0 = {};
  v8f acc1 = {};
  for (int k0 = 0; k0 < H_; k0 += 64) {
    v16bf a0 = ld_b16_split(vbf + k0 + asel, vbf + k0 + 16 + asel);
    v16bf b0 = ld_b16_split(rw + k0 + asel, rw + k0 + 16 + asel);
    acc0 = __builtin_amdgcn_wmma_f32_16x16x32_bf16(false, a0, false, b0, (short)0, acc0, false, false);
    v16bf a1 = ld_b16_split(vbf + k0 + 32 + asel, vbf + k0 + 48 + asel);
    v16bf b1 = ld_b16_split(rw + k0 + 32 + asel, rw + k0 + 48 + asel);
    acc1 = __builtin_amdgcn_wmma_f32_16x16x32_bf16(false, a1, false, b1, (short)0, acc1, false, false);
    prefetch_wgp(rw + k0 + 1024);
  }
  if (lane < 16) dst[nbase + lane] = acc0[0] + acc1[0];
}

// one LSTMCell step: g = x@Wih.T + h@Whh.T ; gates ; h,c,hbf updated
__device__ void lstm_step(unsigned* bar, float* g, float* h, float* c, unsigned short* hbf,
                          const unsigned short* xbf, const unsigned short* wihb,
                          const unsigned short* whhb, const float* b_ih, const float* b_hh) {
  gemv_dual_wmma(g, xbf, hbf, wihb, whhb);
  grid_sync(bar);
  int gt = blockIdx.x * NTHREADS + threadIdx.x;
  if (gt < H_) {
    int j = gt;
    float gi = g[j]          + b_ih[j]          + b_hh[j];
    float gf = g[H_ + j]     + b_ih[H_ + j]     + b_hh[H_ + j];
    float gg = g[2 * H_ + j] + b_ih[2 * H_ + j] + b_hh[2 * H_ + j];
    float go = g[3 * H_ + j] + b_ih[3 * H_ + j] + b_hh[3 * H_ + j];
    float cn = sigmoidf_(gf) * c[j] + sigmoidf_(gi) * tanhf(gg);
    float hn = sigmoidf_(go) * tanhf(cn);
    c[j] = cn; h[j] = hn; hbf[j] = f2bf(hn);
  }
  grid_sync(bar);
}

// ---------------- kernels ----------------
__global__ void prep_kernel(const float* emb, const float* wih, const float* whh,
                            const float* w1, const float* w2, unsigned char* ws) {
  size_t gt = (size_t)blockIdx.x * blockDim.x + threadIdx.x;
  size_t gs = (size_t)gridDim.x * blockDim.x;
  if (gt == 0) {
    unsigned* bar = (unsigned*)(ws + OFF_BAR);
    float* scal = (float*)(ws + OFF_SCAL);
    bar[0] = 0; bar[1] = 0; scal[0] = 0.0f; scal[1] = 0.0f;
  }
  unsigned short* wihb = (unsigned short*)(ws + OFF_WIH);
  unsigned short* whhb = (unsigned short*)(ws + OFF_WHH);
  unsigned short* w1b  = (unsigned short*)(ws + OFF_W1);
  unsigned short* w2b  = (unsigned short*)(ws + OFF_W2);
  unsigned short* embb = (unsigned short*)(ws + OFF_EMBBF);
  for (size_t i = gt; i < (size_t)G4_ * H_; i += gs) wihb[i] = f2bf(wih[i]);
  for (size_t i = gt; i < (size_t)G4_ * H_; i += gs) whhb[i] = f2bf(whh[i]);
  for (size_t i = gt; i < (size_t)H_ * H_;  i += gs) w1b[i]  = f2bf(w1[i]);
  for (size_t i = gt; i < (size_t)H_ * H_;  i += gs) w2b[i]  = f2bf(w2[i]);
  for (size_t i = gt; i < (size_t)(NOPS_ + 1) * H_; i += gs) embb[i] = f2bf(emb[i]);
}

__global__ void __launch_bounds__(NTHREADS)
ctrl_kernel(const float* b_ih, const float* b_hh, const float* w_opt, const float* b_opt,
            const float* v_attn, const int* seedp, float* out, unsigned char* ws) {
  __shared__ float sred[NTHREADS];
  __shared__ float s_sc[16];
  __shared__ int   s_sel;

  unsigned*       bar   = (unsigned*)(ws + OFF_BAR);
  float*          scal  = (float*)(ws + OFF_SCAL);      // [0]=lp, [1]=ent
  float*          stage = (float*)(ws + OFF_STAGE);
  float*          g     = (float*)(ws + OFF_G);
  float*          h     = (float*)(ws + OFF_H);
  float*          c     = (float*)(ws + OFF_C);
  float*          t     = (float*)(ws + OFF_T);
  float*          anch  = (float*)(ws + OFF_ANCH);
  float*          aw1   = (float*)(ws + OFF_AW1);
  unsigned short* xbf   = (unsigned short*)(ws + OFF_XBF);
  unsigned short* hbf   = (unsigned short*)(ws + OFF_HBF);
  unsigned short* embb  = (unsigned short*)(ws + OFF_EMBBF);
  unsigned short* wihb  = (unsigned short*)(ws + OFF_WIH);
  unsigned short* whhb  = (unsigned short*)(ws + OFF_WHH);
  unsigned short* w1b   = (unsigned short*)(ws + OFF_W1);
  unsigned short* w2b   = (unsigned short*)(ws + OFF_W2);

  const int gt = blockIdx.x * NTHREADS + threadIdx.x;
  const int gs = NBLOCKS * NTHREADS;
  const unsigned seedv = (unsigned)seedp[0];

  for (int cell = 0; cell < 2; ++cell) {
    // per-cell init: h,c zero on first cell (hc carried into second), anchors zeroed, x = emb[0]
    if (cell == 0) {
      for (int j = gt; j < H_; j += gs) { h[j] = 0.0f; c[j] = 0.0f; hbf[j] = 0; }
    }
    for (int j = gt; j < NANCH_ * H_; j += gs) { anch[j] = 0.0f; aw1[j] = 0.0f; }
    if (blockIdx.x == 0)
      for (int j = threadIdx.x; j < H_; j += NTHREADS) xbf[j] = embb[j];
    grid_sync(bar);

    // two warmup steps: anchors stay zero, anchors_w1[s] = h@w1.T
    for (int s = 0; s < 2; ++s) {
      lstm_step(bar, g, h, c, hbf, xbf, wihb, whhb, b_ih, b_hh);
      gemv_single_wmma(aw1 + (size_t)s * H_, hbf, w1b, H_);
      grid_sync(bar);
    }

    for (int node = 0; node < NCMP_; ++node) {
      const int s = node + 2;
      // ---- two anchor-index samples ----
      for (int sub = 0; sub < 2; ++sub) {
        lstm_step(bar, g, h, c, hbf, xbf, wihb, whhb, b_ih, b_hh);
        gemv_single_wmma(t, hbf, w2b, H_);   // t = h @ w_attn2.T
        grid_sync(bar);
        if (blockIdx.x == 0) {
          for (int a = 0; a < NANCH_; ++a) {
            float p = 0.0f;
            for (int j = threadIdx.x; j < H_; j += NTHREADS)
              p += tanhf(t[j] + aw1[(size_t)a * H_ + j]) * v_attn[j];
            float r = block_reduce(p, sred);
            if (threadIdx.x == 0) s_sc[a] = r;
          }
          __syncthreads();
          if (threadIdx.x == 0) {
            float lg[NANCH_];
            for (int a = 0; a < NANCH_; ++a) lg[a] = (a < s) ? s_sc[a] : NEGF;
            unsigned ks = mix32(seedv ^ ((unsigned)cell * 0x9E3779B9u) ^
                                ((unsigned)s * 0x85EBCA6Bu) ^ ((unsigned)sub * 0xC2B2AE35u));
            int best = 0; float bv = -3.4e38f;
            for (int a = 0; a < s; ++a) {
              float pv = lg[a] + gumbel_from(mix32(ks ^ (unsigned)(a * 0x27D4EB2F)));
              if (pv > bv) { bv = pv; best = a; }
            }
            float mx = lg[0];
            for (int a = 1; a < NANCH_; ++a) mx = fmaxf(mx, lg[a]);
            float se = 0.0f;
            for (int a = 0; a < NANCH_; ++a) se += __expf(lg[a] - mx);
            float lse = mx + logf(se);
            scal[0] += (lse - lg[best]);          // lp += -logp[idx]
            float ent = 0.0f;
            for (int a = 0; a < NANCH_; ++a) {
              if (lg[a] > 0.5f * NEGF) { float l = lg[a] - lse; ent -= __expf(l) * l; }
            }
            scal[1] += ent;
            stage[cell * 24 + node * 2 + sub] = (float)best;
            s_sel = best;
          }
          __syncthreads();
          const int pick = s_sel;
          for (int j = threadIdx.x; j < H_; j += NTHREADS)
            xbf[j] = f2bf(anch[(size_t)pick * H_ + j]);   // x = anchors[idx]
        }
        grid_sync(bar);
      }
      // ---- two op samples ----
      for (int sub = 0; sub < 2; ++sub) {
        lstm_step(bar, g, h, c, hbf, xbf, wihb, whhb, b_ih, b_hh);
        if (blockIdx.x == 0) {
          for (int o = 0; o < NOPS_; ++o) {
            float p = 0.0f;
            for (int j = threadIdx.x; j < H_; j += NTHREADS)
              p += h[j] * w_opt[(size_t)o * H_ + j];
            float r = block_reduce(p, sred);
            if (threadIdx.x == 0) s_sc[o] = r + b_opt[o];
          }
          __syncthreads();
          if (threadIdx.x == 0) {
            unsigned ks = mix32(seedv ^ ((unsigned)cell * 0x9E3779B9u) ^
                                ((unsigned)s * 0x85EBCA6Bu) ^ ((unsigned)(2 + sub) * 0xC2B2AE35u));
            int best = 0; float bv = -3.4e38f;
            for (int o = 0; o < NOPS_; ++o) {
              float pv = s_sc[o] + gumbel_from(mix32(ks ^ (unsigned)(o * 0x27D4EB2F)));
              if (pv > bv) { bv = pv; best = o; }
            }
            float mx = s_sc[0];
            for (int o = 1; o < NOPS_; ++o) mx = fmaxf(mx, s_sc[o]);
            float se = 0.0f;
            for (int o = 0; o < NOPS_; ++o) se += __expf(s_sc[o] - mx);
            float lse = mx + logf(se);
            scal[0] += (lse - s_sc[best]);
            float ent = 0.0f;
            for (int o = 0; o < NOPS_; ++o) { float l = s_sc[o] - lse; ent -= __expf(l) * l; }
            scal[1] += ent;
            stage[cell * 24 + 12 + node * 2 + sub] = (float)best;
            s_sel = best;
          }
          __syncthreads();
          const int op = s_sel;
          for (int j = threadIdx.x; j < H_; j += NTHREADS)
            xbf[j] = embb[(size_t)(op + 1) * H_ + j];     // x = emb[op+1]
        }
        grid_sync(bar);
      }
      // ---- anchor update ----
      lstm_step(bar, g, h, c, hbf, xbf, wihb, whhb, b_ih, b_hh);
      for (int j = gt; j < H_; j += gs) anch[(size_t)s * H_ + j] = h[j];
      gemv_single_wmma(aw1 + (size_t)s * H_, hbf, w1b, H_);
      if (blockIdx.x == 0)
        for (int j = threadIdx.x; j < H_; j += NTHREADS) xbf[j] = embb[j];  // x = emb[0]
      grid_sync(bar);
    }
  }

  // outputs: idx_n[6,2], opt_n[6,2], idx_r[6,2], opt_r[6,2], ent, lp  (50 floats)
  if (blockIdx.x == 0) {
    int tt = threadIdx.x;
    if (tt < 48) out[tt] = stage[tt];
    if (tt == 48) out[48] = scal[1];
    if (tt == 49) out[49] = scal[0];
  }
}

extern "C" void kernel_launch(void* const* d_in, const int* in_sizes, int n_in,
                              void* d_out, int out_size, void* d_ws, size_t ws_size,
                              hipStream_t stream) {
  (void)in_sizes; (void)n_in; (void)out_size; (void)ws_size;
  const float* emb    = (const float*)d_in[0];
  const float* w_ih   = (const float*)d_in[1];
  const float* w_hh   = (const float*)d_in[2];
  const float* b_ih   = (const float*)d_in[3];
  const float* b_hh   = (const float*)d_in[4];
  const float* w_opt  = (const float*)d_in[5];
  const float* b_opt  = (const float*)d_in[6];
  const float* w1     = (const float*)d_in[7];
  const float* w2     = (const float*)d_in[8];
  const float* v_attn = (const float*)d_in[9];
  const int*   seed   = (const int*)d_in[10];
  unsigned char* ws = (unsigned char*)d_ws;
  float* out = (float*)d_out;

  prep_kernel<<<2048, 256, 0, stream>>>(emb, w_ih, w_hh, w1, w2, ws);
  ctrl_kernel<<<NBLOCKS, NTHREADS, 0, stream>>>(b_ih, b_hh, w_opt, b_opt, v_attn, seed, out, ws);
}
